// ConvLSTM_78297253806644
// MI455X (gfx1250) — matrix operand order repaired and checked by reading, not compiled, not run on hardware
//
#include <hip/hip_runtime.h>
#include <hip/hip_bf16.h>
#include <math.h>

typedef __attribute__((ext_vector_type(16))) _Float16 v16h;
typedef __attribute__((ext_vector_type(8)))  _Float16 v8h;
typedef __attribute__((ext_vector_type(8)))  float    v8f;
typedef __attribute__((ext_vector_type(4)))  unsigned int v4u;
typedef __attribute__((ext_vector_type(8)))  int v8i;
typedef __attribute__((ext_vector_type(4)))  int v4i;

#define WMMA_F16(a,b,c) __builtin_amdgcn_wmma_f32_16x16x32_f16(false,(a),false,(b),(short)0,(c),false,false)

#define VOX 512   // 8*8*8 output voxels
#define HD  64
#define GD  256   // 4*H
#define SL  7     // timesteps
#define NS  2     // N*C samples
// padded activation geometry: [10][10][10][64] f16 per sample
#define PY  640         // padded y stride (10*64)
#define PZ  6400        // padded z stride (10*640)
#define PS  64000       // padded sample stride (10*6400)
#define STAGE_ELEMS (3*4*PY)   // 7680 f16 = 15360 B halo tile per block

// ---- Tensor Data Mover availability (probe-verified builtins; guarded) -----
#if defined(__has_builtin)
#  if __has_builtin(__builtin_amdgcn_tensor_load_to_lds) && __has_builtin(__builtin_amdgcn_s_wait_tensorcnt)
#    define USE_TDM 1
#  else
#    define USE_TDM 0
#  endif
#else
#  define USE_TDM 0
#endif
#if defined(__has_include)
#  if __has_include(<hip/amd_detail/amd_gfx1250_TDM.h>)
#    define TDM_6ARG 1
#  else
#    define TDM_6ARG 0
#  endif
#else
#  define TDM_6ARG 0
#endif

__device__ inline v8f zero8(){ v8f z; for(int i=0;i<8;++i) z[i]=0.f; return z; }

// A fragment for 16x16x32 f16 WMMA: lane holds M=lane%16, K chunks
// [base,base+8) and [base+16,base+24), base=(lane/16)*8 — two 16B loads.
__device__ inline v16h load_a16(const _Float16* p){
  union { v16h v; v8h h[2]; } u;
  u.h[0] = *(const v8h*)(p);
  u.h[1] = *(const v8h*)(p + 16);
  return u.v;
}

__device__ inline float sigm(float x){ return 1.f/(1.f+expf(-x)); }

// Stage the 3z x 4y x 640 halo tile (15KB) of a padded sample into LDS.
// TDM path: one tensor_load_to_lds (D# per CDNA5 ISA 8.3-8.6), issued by wave 0.
__device__ inline void stage_tile(const _Float16* src, _Float16* stage, unsigned tid){
#if USE_TDM
  if ((tid >> 5) == 0){
    unsigned long long ga = (unsigned long long)src;        // byte address
    unsigned la = (unsigned)(unsigned long long)stage;      // LDS byte offset (addr[31:0])
    v4u g0;
    g0[0] = 1u;                                   // count=1, user descriptor
    g0[1] = la;                                   // lds_addr
    g0[2] = (unsigned)ga;                         // global_addr[31:0]
    g0[3] = (unsigned)((ga >> 32) & 0x01FFFFFFull) | (2u << 30);  // addr[56:32] | type=2
    v8i g1;
    g1[0] = 0x00010000;        // workgroup_mask=0, data_size=1 (2 bytes)
    g1[1] = (PY << 16);        // tensor_dim0[15:0]=640
    g1[2] = (10 << 16);        // tensor_dim0 hi=0, tensor_dim1[15:0]=10
    g1[3] = (PY << 16);        // tensor_dim1 hi=0, tile_dim0=640
    g1[4] = 4 | (3 << 16);     // tile_dim1=4 (y rows), tile_dim2=3 (z planes)
    g1[5] = PY;                // tensor_dim0_stride lo32 = 640
    g1[6] = (PZ << 16);        // stride0 hi=0, tensor_dim1_stride[15:0]=6400
    g1[7] = 0;                 // stride1 hi
    v4i g2; g2[0] = 10; g2[1] = 0; g2[2] = PZ; g2[3] = 0;  // tensor_dim2=10, dim2_stride=6400
    v4i g3; g3[0] = 0; g3[1] = 0; g3[2] = 0; g3[3] = 0;
#if TDM_6ARG
    v8i g4; for (int i = 0; i < 8; ++i) g4[i] = 0;
    __builtin_amdgcn_tensor_load_to_lds(g0, g1, g2, g3, g4, 0);
#else
    __builtin_amdgcn_tensor_load_to_lds(g0, g1, g2, g3, 0);
#endif
    __builtin_amdgcn_s_wait_tensorcnt(0);
  }
#else
  for (int i = tid; i < STAGE_ELEMS/8; i += 256){   // 960 16B chunks
    int zi = i / 320; int r = i - zi*320;
    int yi = r / 80;  int xi = r - yi*80;
    *(v8h*)(stage + i*8) = *(const v8h*)(src + zi*PZ + yi*PY + xi*8);
  }
#endif
  __syncthreads();
}

// -------- pack f32 [K][N] row-major weights into WMMA B-fragment order ------
// tiles (kt,nt) of 512 f16; lane L holds col n=nt*16+L%16, k=kt*32+(L/16)*8+(e%8)+(e/8)*16.
__global__ void k_pack_b(const float* __restrict__ src, _Float16* __restrict__ dst,
                         int K, int Nn, int total){
  int t = blockIdx.x*256 + threadIdx.x;
  if (t >= total) return;
  int NT = Nn >> 4;
  int e    = t & 15;
  int lane = (t >> 4) & 31;
  int tile = t >> 9;
  int kt = tile / NT;
  int nt = tile - kt*NT;
  int n = (nt << 4) + (lane & 15);
  int k = (kt << 5) + ((lane >> 4) << 3) + (e & 7) + ((e >> 3) << 4);
  dst[t] = (_Float16)src[(size_t)k*Nn + n];
}

// --------------- embedding gather into padded h: [n][s][10][10][10][64] -----
__global__ void k_embed(const int* __restrict__ code, const int* __restrict__ ncode,
                        const float* __restrict__ emb, _Float16* __restrict__ hpad,
                        int total){
  int t = blockIdx.x*256 + threadIdx.x;
  if (t >= total) return;
  int c = t & 63;
  int rest = t >> 6;
  int v = rest & 511;
  int rest2 = rest >> 9;
  int s = rest2 % SL;
  int n = rest2 / SL;
  int id = (s < 5) ? code[(n*5 + s)*VOX + v] : ncode[(n*3 + (s-5))*VOX + v];
  int z = v >> 6, y = (v >> 3) & 7, x = v & 7;
  hpad[(size_t)(n*SL + s)*PS + (z+1)*PZ + (y+1)*PY + (x+1)*64 + c] = (_Float16)emb[id*HD + c];
}

__global__ void k_zero(float* __restrict__ p, int total){
  int t = blockIdx.x*256 + threadIdx.x;
  if (t < total) p[t] = 0.f;
}

// ---- shared GEMM core: 16 voxels x 256 ch, K=54x32, A from LDS halo tile ---
// Caller provides abase (lane A pointer in stage) and bbase (lane B pointer).
#define CONV_GEMM_BODY(abase, bbase, acc0, acc1)                              \
  _Pragma("unroll")                                                           \
  for (int dz = 0; dz < 3; ++dz)                                              \
  _Pragma("unroll")                                                           \
  for (int dy = 0; dy < 3; ++dy)                                              \
  _Pragma("unroll")                                                           \
  for (int dx = 0; dx < 3; ++dx)                                              \
  _Pragma("unroll")                                                           \
  for (int kh = 0; kh < 2; ++kh){                                             \
    int kt = (dz*9 + dy*3 + dx)*2 + kh;                                       \
    v16h a = load_a16((abase) + (dz*4 + dy)*PY + dx*64 + kh*32);              \
    const _Float16* pb = (bbase) + (size_t)kt*8192;  /* 16 tiles * 512 */     \
    v16h b0 = *(const v16h*)pb;                                               \
    v16h b1 = *(const v16h*)(pb + 512);                                       \
    acc0 = WMMA_F16(a, b0, acc0);                                             \
    acc1 = WMMA_F16(a, b1, acc1);                                             \
  }

// -------- input-path conv: gi = LN(conv3d(h)+bin) for all 14 (n,s) ----------
__global__ __launch_bounds__(256) void k_conv_gates_in(
    const _Float16* __restrict__ hpad, const _Float16* __restrict__ bp,
    const float* __restrict__ bin, const float* __restrict__ gw,
    const float* __restrict__ gb, float* __restrict__ gi)
{
  __shared__ __align__(32) _Float16 stage[STAGE_ELEMS];
  __shared__ float tile[16*GD];
  __shared__ float mu[16], rs[16];
  int p = blockIdx.y;                  // 0..13
  int n = p / SL, s = p - n*SL;
  int mtile = blockIdx.x;              // 0..31 : z=mtile/4, y-pair=(mtile%4)*2
  int Z = mtile >> 2, Y = (mtile & 3) << 1;
  stage_tile(hpad + (size_t)(n*SL + s)*PS + Z*PZ + Y*PY, stage, threadIdx.x);

  int lane = threadIdx.x & 31, wave = threadIdx.x >> 5;
  int m = lane & 15, half = lane >> 4;
  int ly = m >> 3, xc = m & 7;
  const _Float16* abase = stage + ly*PY + xc*64 + half*8;
  const _Float16* bbase = bp + (size_t)(wave*2)*512 + (lane << 4);
  int nt0 = wave*2, nt1 = nt0 + 1;
  v8f acc0 = zero8(), acc1 = zero8();
  CONV_GEMM_BODY(abase, bbase, acc0, acc1)

  int nn = lane & 15;
  #pragma unroll
  for (int r = 0; r < 8; ++r){
    int M = r + (half << 3);
    tile[M*GD + (nt0<<4) + nn] = acc0[r] + bin[(nt0<<4)+nn];
    tile[M*GD + (nt1<<4) + nn] = acc1[r] + bin[(nt1<<4)+nn];
  }
  __syncthreads();
  if (threadIdx.x < 16){
    float sum = 0.f;
    for (int c = 0; c < GD; ++c) sum += tile[threadIdx.x*GD + c];
    float mu_ = sum*(1.f/GD);
    float var = 0.f;
    for (int c = 0; c < GD; ++c){ float d = tile[threadIdx.x*GD + c] - mu_; var += d*d; }
    mu[threadIdx.x] = mu_;
    rs[threadIdx.x] = rsqrtf(var*(1.f/GD) + 1e-5f);
  }
  __syncthreads();
  float* gout = gi + (size_t)((s*NS + n)*VOX + mtile*16)*GD;
  for (int idx = threadIdx.x; idx < 16*GD; idx += 256){
    int mv = idx >> 8, ch = idx & 255;
    gout[idx] = (tile[idx] - mu[mv])*rs[mv]*gw[ch] + gb[ch];
  }
}

// -------- recurrent step: gates = gi[s] + LN(conv3d(hx)+bh); LSTM update ----
__global__ __launch_bounds__(256) void k_conv_step(
    const _Float16* __restrict__ hxr, _Float16* __restrict__ hxw,
    const _Float16* __restrict__ bp, const float* __restrict__ bh,
    const float* __restrict__ gw, const float* __restrict__ gb,
    const float* __restrict__ gi, float* __restrict__ cx,
    _Float16* __restrict__ hpad, int s)
{
  __shared__ __align__(32) _Float16 stage[STAGE_ELEMS];
  __shared__ float tile[16*GD];
  __shared__ float mu[16], rs[16];
  int n = blockIdx.y;                  // 0..1
  int mtile = blockIdx.x;
  int Z = mtile >> 2, Y = (mtile & 3) << 1;
  stage_tile(hxr + (size_t)n*PS + Z*PZ + Y*PY, stage, threadIdx.x);

  int lane = threadIdx.x & 31, wave = threadIdx.x >> 5;
  int m = lane & 15, half = lane >> 4;
  int ly = m >> 3, xc = m & 7;
  const _Float16* abase = stage + ly*PY + xc*64 + half*8;
  const _Float16* bbase = bp + (size_t)(wave*2)*512 + (lane << 4);
  int nt0 = wave*2, nt1 = nt0 + 1;
  v8f acc0 = zero8(), acc1 = zero8();
  CONV_GEMM_BODY(abase, bbase, acc0, acc1)

  int nn = lane & 15;
  #pragma unroll
  for (int r = 0; r < 8; ++r){
    int M = r + (half << 3);
    tile[M*GD + (nt0<<4) + nn] = acc0[r] + bh[(nt0<<4)+nn];
    tile[M*GD + (nt1<<4) + nn] = acc1[r] + bh[(nt1<<4)+nn];
  }
  __syncthreads();
  if (threadIdx.x < 16){
    float sum = 0.f;
    for (int c = 0; c < GD; ++c) sum += tile[threadIdx.x*GD + c];
    float mu_ = sum*(1.f/GD);
    float var = 0.f;
    for (int c = 0; c < GD; ++c){ float d = tile[threadIdx.x*GD + c] - mu_; var += d*d; }
    mu[threadIdx.x] = mu_;
    rs[threadIdx.x] = rsqrtf(var*(1.f/GD) + 1e-5f);
  }
  __syncthreads();
  const float* gin = gi + (size_t)((s*NS + n)*VOX)*GD;
  for (int q = threadIdx.x; q < 16*HD; q += 256){
    int mv = q >> 6, c = q & 63;
    int v = mtile*16 + mv;
    const float* gv = gin + (size_t)v*GD;
    float muv = mu[mv], rsv = rs[mv];
    float ig = gv[c]       + (tile[mv*GD + c]       - muv)*rsv*gw[c]       + gb[c];
    float fg = gv[64 + c]  + (tile[mv*GD + 64 + c]  - muv)*rsv*gw[64 + c]  + gb[64 + c];
    float cg = gv[128 + c] + (tile[mv*GD + 128 + c] - muv)*rsv*gw[128 + c] + gb[128 + c];
    float og = gv[192 + c] + (tile[mv*GD + 192 + c] - muv)*rsv*gw[192 + c] + gb[192 + c];
    size_t ci = (size_t)(n*VOX + v)*HD + c;
    float cn = sigm(fg)*cx[ci] + sigm(ig)*tanhf(cg);
    float hn = sigm(og)*tanhf(cn);
    cx[ci] = cn;
    int z = v >> 6, y = (v >> 3) & 7, x = v & 7;
    size_t pe = (size_t)(z+1)*PZ + (y+1)*PY + (x+1)*64 + c;
    hxw[(size_t)n*PS + pe] = (_Float16)hn;
    hpad[(size_t)(n*SL + s)*PS + pe] = (_Float16)hn;
  }
}

// -------- head1: z = gelu(LN(h_last @ w1 + b1)) -----------------------------
__global__ __launch_bounds__(128) void k_head1(
    const _Float16* __restrict__ hpad, const _Float16* __restrict__ bp,
    const float* __restrict__ bias1, const float* __restrict__ lg,
    const float* __restrict__ lb, _Float16* __restrict__ z16)
{
  __shared__ float tile[16*HD];
  __shared__ float mu[16], rs[16];
  int b = blockIdx.x;                  // 0..191
  int n = b / 96; int r = b - n*96;
  int sIdx = r >> 5; int vt = r & 31;
  const _Float16* hb = hpad + (size_t)(n*SL + 4 + sIdx)*PS;
  int lane = threadIdx.x & 31, wave = threadIdx.x >> 5;   // 4 waves, 4 N-tiles
  int m = lane & 15, half = lane >> 4;
  int v = vt*16 + m;
  int z = v >> 6, y = (v >> 3) & 7, x = v & 7;
  const _Float16* arow = hb + (size_t)(z+1)*PZ + (y+1)*PY + (x+1)*64;
  v8f acc = zero8();
  #pragma unroll
  for (int kt = 0; kt < 2; ++kt){
    int c0 = (kt << 5) + (half << 3);
    v16h a = load_a16(arow + c0);
    v16h bf = *(const v16h*)(bp + (((size_t)(kt*4 + wave)) << 9) + (lane << 4));
    acc = WMMA_F16(a, bf, acc);
  }
  int nn = lane & 15;
  #pragma unroll
  for (int rr = 0; rr < 8; ++rr){
    int M = rr + (half << 3);
    tile[M*HD + (wave<<4) + nn] = acc[rr] + bias1[(wave<<4)+nn];
  }
  __syncthreads();
  if (threadIdx.x < 16){
    float sum = 0.f;
    for (int c = 0; c < HD; ++c) sum += tile[threadIdx.x*HD + c];
    float mu_ = sum*(1.f/HD);
    float var = 0.f;
    for (int c = 0; c < HD; ++c){ float d = tile[threadIdx.x*HD + c] - mu_; var += d*d; }
    mu[threadIdx.x] = mu_;
    rs[threadIdx.x] = rsqrtf(var*(1.f/HD) + 1e-5f);
  }
  __syncthreads();
  _Float16* zo = z16 + ((size_t)(n*3 + sIdx)*VOX + vt*16)*HD;
  for (int idx = threadIdx.x; idx < 16*HD; idx += 128){
    int mv = idx >> 6, ch = idx & 63;
    float xv = (tile[idx] - mu[mv])*rs[mv]*lg[ch] + lb[ch];
    zo[idx] = (_Float16)(0.5f*xv*(1.0f + erff(xv*0.70710678118654752f)));
  }
}

// -------- head2: logits = z @ w2 + b2; also write transposed score ----------
__global__ __launch_bounds__(256) void k_head2(
    const _Float16* __restrict__ z16, const _Float16* __restrict__ bp,
    const float* __restrict__ b2, float* __restrict__ logits,
    float* __restrict__ score)
{
  int blk = blockIdx.x;                 // 192 blocks of 16 rows
  int lane = threadIdx.x & 31, wave = threadIdx.x >> 5;  // 8 waves x 4 N-tiles
  int m = lane & 15, half = lane >> 4;
  const _Float16* zb = z16 + (size_t)blk*16*HD;
  v8f acc[4];
  for (int i = 0; i < 4; ++i) acc[i] = zero8();
  #pragma unroll
  for (int kt = 0; kt < 2; ++kt){
    int c0 = (kt << 5) + (half << 3);
    v16h a = load_a16(zb + m*HD + c0);
    #pragma unroll
    for (int i = 0; i < 4; ++i){
      int nt = wave*4 + i;
      v16h bf = *(const v16h*)(bp + (((size_t)(kt*32 + nt)) << 9) + (lane << 4));
      acc[i] = WMMA_F16(a, bf, acc[i]);
    }
  }
  int nn = lane & 15;
  #pragma unroll
  for (int i = 0; i < 4; ++i){
    int voc = (wave*4 + i)*16 + nn;
    #pragma unroll
    for (int r = 0; r < 8; ++r){
      int M = r + (half << 3);
      int row = blk*16 + M;
      float val = acc[i][r] + b2[voc];
      logits[(size_t)row*512 + voc] = val;
      int n2 = row / 1536; int rr = row - n2*1536;
      int sIdx = rr >> 9; int v = rr & 511;
      score[(((size_t)n2*512 + voc)*3 + sIdx)*VOX + v] = val;
    }
  }
}

// -------- per-row softmax: loss (atomicAdd) + argmax pred -------------------
__global__ __launch_bounds__(256) void k_loss_pred(
    const float* __restrict__ logits, const int* __restrict__ ncode,
    float* __restrict__ d_loss, float* __restrict__ d_pred)
{
  __shared__ float smax[256]; __shared__ int sidx[256]; __shared__ float ssum[256];
  int row = blockIdx.x, t = threadIdx.x;
  const float* lr = logits + (size_t)row*512;
  float mv = lr[t]; int mi = t;
  float o = lr[t + 256];
  if (o > mv){ mv = o; mi = t + 256; }
  smax[t] = mv; sidx[t] = mi;
  __syncthreads();
  for (int off = 128; off; off >>= 1){
    if (t < off){
      float ov = smax[t + off]; int oi = sidx[t + off];
      if (ov > smax[t] || (ov == smax[t] && oi < sidx[t])){ smax[t] = ov; sidx[t] = oi; }
    }
    __syncthreads();
  }
  float gmax = smax[0]; int gidx = sidx[0];
  float e = expf(lr[t] - gmax) + expf(lr[t + 256] - gmax);
  ssum[t] = e;
  __syncthreads();
  for (int off = 128; off; off >>= 1){
    if (t < off) ssum[t] += ssum[t + off];
    __syncthreads();
  }
  if (t == 0){
    int tgt = ncode[row];
    float lp = lr[tgt] - gmax - logf(ssum[0]);
    atomicAdd(d_loss, -lp * (1.0f/3072.0f));
    d_pred[row] = (float)gidx;
  }
}

extern "C" void kernel_launch(void* const* d_in, const int* in_sizes, int n_in,
                              void* d_out, int out_size, void* d_ws, size_t ws_size,
                              hipStream_t stream)
{
  const int*   code  = (const int*)d_in[0];
  const int*   ncode = (const int*)d_in[1];
  const float* emb   = (const float*)d_in[2];
  const float* win   = (const float*)d_in[3];
  const float* bin_  = (const float*)d_in[4];
  const float* gin_w = (const float*)d_in[5];
  const float* gin_b = (const float*)d_in[6];
  const float* wh    = (const float*)d_in[7];
  const float* bh    = (const float*)d_in[8];
  const float* gh_w  = (const float*)d_in[9];
  const float* gh_b  = (const float*)d_in[10];
  const float* w1    = (const float*)d_in[11];
  const float* b1    = (const float*)d_in[12];
  const float* ln_g  = (const float*)d_in[13];
  const float* ln_b  = (const float*)d_in[14];
  const float* w2    = (const float*)d_in[15];
  const float* b2    = (const float*)d_in[16];

  char* pw = (char*)d_ws;
  _Float16* hpad = (_Float16*)pw; pw += (size_t)NS*SL*PS*2;     // 1,792,000 B padded h
  float*    gi   = (float*)pw;    pw += (size_t)SL*NS*VOX*GD*4; // 7,340,032 B
  float*    cx   = (float*)pw;    pw += (size_t)NS*VOX*HD*4;    //   262,144 B
  _Float16* hx0  = (_Float16*)pw; pw += (size_t)NS*PS*2;        //   256,000 B padded hx
  _Float16* hx1  = (_Float16*)pw; pw += (size_t)NS*PS*2;
  _Float16* wpki[2]; _Float16* wpkh[2];
  for (int l = 0; l < 2; ++l){ wpki[l] = (_Float16*)pw; pw += (size_t)1728*256*2; }
  for (int l = 0; l < 2; ++l){ wpkh[l] = (_Float16*)pw; pw += (size_t)1728*256*2; }
  _Float16* wpk1 = (_Float16*)pw; pw += (size_t)64*64*2;
  _Float16* wpk2 = (_Float16*)pw; pw += (size_t)64*512*2;
  _Float16* z16  = (_Float16*)pw; pw += (size_t)NS*3*VOX*HD*2;
  float* logits  = (float*)pw;    pw += (size_t)NS*3*VOX*512*4;

  float* out    = (float*)d_out;
  float* score  = out;                       // 2*512*3*1*512 = 1,572,864
  float* d_loss = out + 1572864;
  float* d_pred = out + 1572865;             // 3072 preds (as float)

  auto nb = [](int total){ return (total + 255) / 256; };

  // weight packing (deterministic every call)
  for (int l = 0; l < 2; ++l){
    k_pack_b<<<nb(442368), 256, 0, stream>>>(win + (size_t)l*442368, wpki[l], 1728, 256, 442368);
    k_pack_b<<<nb(442368), 256, 0, stream>>>(wh  + (size_t)l*442368, wpkh[l], 1728, 256, 442368);
  }
  k_pack_b<<<nb(4096),  256, 0, stream>>>(w1, wpk1, 64, 64,  4096);
  k_pack_b<<<nb(32768), 256, 0, stream>>>(w2, wpk2, 64, 512, 32768);

  // zero padded activation buffer (borders stay zero), then embed interior
  k_zero<<<nb((int)(NS*SL*PS/2)), 256, 0, stream>>>((float*)hpad, (int)(NS*SL*PS/2));
  k_embed<<<nb(NS*SL*VOX*HD), 256, 0, stream>>>(code, ncode, emb, hpad, NS*SL*VOX*HD);
  k_zero<<<1, 1, 0, stream>>>(d_loss, 1);

  for (int l = 0; l < 2; ++l){
    k_conv_gates_in<<<dim3(32, NS*SL), 256, 0, stream>>>(
        hpad, wpki[l], bin_ + l*GD, gin_w + l*GD, gin_b + l*GD, gi);
    k_zero<<<nb(NS*VOX*HD), 256, 0, stream>>>(cx, NS*VOX*HD);
    k_zero<<<nb((int)(NS*PS/2)), 256, 0, stream>>>((float*)hx0, (int)(NS*PS/2));
    k_zero<<<nb((int)(NS*PS/2)), 256, 0, stream>>>((float*)hx1, (int)(NS*PS/2));
    for (int s = 0; s < SL; ++s){
      const _Float16* hr = (s & 1) ? hx1 : hx0;
      _Float16*       hw = (s & 1) ? hx0 : hx1;
      k_conv_step<<<dim3(32, NS), 256, 0, stream>>>(
          hr, hw, wpkh[l], bh + l*GD, gh_w + l*GD, gh_b + l*GD,
          gi, cx, hpad, s);
    }
  }

  k_head1<<<192, 128, 0, stream>>>(hpad, wpk1, b1, ln_g, ln_b, z16);
  k_head2<<<192, 256, 0, stream>>>(z16, wpk2, b2, logits, score);
  k_loss_pred<<<3072, 256, 0, stream>>>(logits, ncode, d_loss, d_pred);
}
